// MultiHeadedAttention_5128190951579
// MI455X (gfx1250) — matrix operand order repaired and checked by reading, not compiled, not run on hardware
//
#include <hip/hip_runtime.h>

// ---------------------------------------------------------------------------
// MI455X (gfx1250) multi-head attention, f16 WMMA (v_wmma_f32_16x16x32_f16)
// ---------------------------------------------------------------------------

#define DEV __device__ __forceinline__

typedef _Float16 v8h  __attribute__((ext_vector_type(8)));
typedef _Float16 v16h __attribute__((ext_vector_type(16)));
typedef float    v8f  __attribute__((ext_vector_type(8)));
typedef float    v4f  __attribute__((ext_vector_type(4)));

constexpr int CB  = 2;     // batch
constexpr int LQ  = 2048;
constexpr int LKV = 4096;
constexpr int CD  = 512;   // model dim
constexpr int CH  = 8;     // heads
constexpr int DK  = 64;    // head dim

DEV v8f wmma16(v16h a, v16h b, v8f c) {
  return __builtin_amdgcn_wmma_f32_16x16x32_f16(false, a, false, b, (short)0, c,
                                                false, false);
}

// ds_swizzle BITMASK_PERM: offset[4:0]=and(0x1f), [9:5]=or(0), [14:10]=xor(m)
// xor masks <= 8 stay within 16-lane groups (matches C/D row layout halves).
#define SWZ_XOR(v, m)                                                          \
  __int_as_float(__builtin_amdgcn_ds_swizzle(__float_as_int(v),                \
                                             (0x001F | ((m) << 10))))

DEV float redmax16(float v) {
  v = fmaxf(v, SWZ_XOR(v, 1));
  v = fmaxf(v, SWZ_XOR(v, 2));
  v = fmaxf(v, SWZ_XOR(v, 4));
  v = fmaxf(v, SWZ_XOR(v, 8));
  return v;
}
DEV float redsum16(float v) {
  v += SWZ_XOR(v, 1);
  v += SWZ_XOR(v, 2);
  v += SWZ_XOR(v, 4);
  v += SWZ_XOR(v, 8);
  return v;
}

DEV v16h pack16(v8h lo, v8h hi) {
  v16h r;
#pragma unroll
  for (int i = 0; i < 8; ++i) { r[i] = lo[i]; r[8 + i] = hi[i]; }
  return r;
}

// --------------------------- weight transpose ------------------------------
// WT[n][k] = (f16) W[k][n]   (512x512)
__global__ __launch_bounds__(256) void wt_kernel(const float* __restrict__ W,
                                                 _Float16* __restrict__ WT) {
  int i = blockIdx.x * 256 + threadIdx.x;   // over 512*512
  int n = i >> 9, k = i & 511;
  WT[i] = (_Float16)W[(size_t)k * CD + n];
}

// ------------------------------ projections --------------------------------
// C[M x 512] = A[M x 512] @ W + bias, using WT (f16, transposed) as B operand.
// MODE 0: out f16 row-major, scaled by 1/sqrt(DK)   (Q)
// MODE 1: out f16 row-major                         (K)
// MODE 2: out f16 transposed per batch Vt[b][d][kv] (V)
// MODE 3: out f32 row-major                         (final)
template <int MODE, bool AF16>
__global__ __launch_bounds__(256) void proj_kernel(
    const void* __restrict__ Av, const _Float16* __restrict__ WT,
    const float* __restrict__ bias, void* __restrict__ Outv, int M) {
  const int lane = threadIdx.x & 31;
  const int wid  = blockIdx.x * 8 + (threadIdx.x >> 5);
  const int n0   = (wid & 7) * 64;         // N/64 = 8 tiles
  const int m0   = (wid >> 3) * 16;
  const int grp  = lane >> 4;
  const int ln   = lane & 15;
  (void)M;

  v8f acc[4] = {};
  for (int k0 = 0; k0 < CD; k0 += 32) {
    v16h a;
    if (AF16) {
      const _Float16* ar = (const _Float16*)Av + (size_t)(m0 + ln) * CD + k0;
      a = pack16(*(const v8h*)(ar + 16 * grp), *(const v8h*)(ar + 8 + 16 * grp));
    } else {
      const float* ar = (const float*)Av + (size_t)(m0 + ln) * CD + k0;
      v4f f0 = *(const v4f*)(ar + 16 * grp);
      v4f f1 = *(const v4f*)(ar + 16 * grp + 4);
      v4f f2 = *(const v4f*)(ar + 8 + 16 * grp);
      v4f f3 = *(const v4f*)(ar + 8 + 16 * grp + 4);
#pragma unroll
      for (int i = 0; i < 4; ++i) {
        a[i]      = (_Float16)f0[i];
        a[4 + i]  = (_Float16)f1[i];
        a[8 + i]  = (_Float16)f2[i];
        a[12 + i] = (_Float16)f3[i];
      }
    }
#pragma unroll
    for (int j = 0; j < 4; ++j) {
      const _Float16* wr =
          WT + (size_t)(n0 + 16 * j + ln) * CD + k0 + 16 * grp;
      acc[j] = wmma16(a, *(const v16h*)wr, acc[j]);
    }
  }

#pragma unroll
  for (int j = 0; j < 4; ++j) {
    const int col = n0 + 16 * j + ln;
    const float bval = bias[col];
#pragma unroll
    for (int r = 0; r < 8; ++r) {
      const int row = m0 + r + 8 * grp;   // C/D layout: M = r + 8*(lane/16)
      float v = acc[j][r] + bval;
      if (MODE == 0) {
        ((_Float16*)Outv)[(size_t)row * CD + col] = (_Float16)(v * 0.125f);
      } else if (MODE == 1) {
        ((_Float16*)Outv)[(size_t)row * CD + col] = (_Float16)v;
      } else if (MODE == 2) {
        const int bb = row / LKV, kv = row % LKV;
        ((_Float16*)Outv)[((size_t)bb * CD + col) * LKV + kv] = (_Float16)v;
      } else {
        ((float*)Outv)[(size_t)row * CD + col] = v;
      }
    }
  }
}

// ------------------------------- attention ---------------------------------
// One wave per (b, h, 16-row q tile). Flash-attention over kv chunks of 32.
__global__ __launch_bounds__(256) void attn_kernel(
    const _Float16* __restrict__ Q,   // [B*LQ][512], pre-scaled by 1/8
    const _Float16* __restrict__ K,   // [B*LKV][512]
    const _Float16* __restrict__ Vt,  // [B][512][LKV]
    const unsigned char* __restrict__ mask,  // [B][LKV], 1 = padded
    _Float16* __restrict__ X) {       // [B*LQ][512]
  __shared__ __align__(32) _Float16 plds[8][16 * 32];  // per-wave P tile

  const int lane = threadIdx.x & 31;
  const int wib  = threadIdx.x >> 5;
  const int wid  = blockIdx.x * 8 + wib;
  const int q0   = (wid & 127) * 16;       // LQ/16 = 128
  const int h    = (wid >> 7) & 7;
  const int b    = wid >> 10;
  const int grp  = lane >> 4;
  const int ln   = lane & 15;

  // Q tile as two A-operands (dk chunks 0-31, 32-63)
  const _Float16* qrow = Q + ((size_t)(b * LQ + q0 + ln)) * CD + h * DK;
  v16h qa[2];
#pragma unroll
  for (int c = 0; c < 2; ++c)
    qa[c] = pack16(*(const v8h*)(qrow + 32 * c + 16 * grp),
                   *(const v8h*)(qrow + 32 * c + 8 + 16 * grp));

  v8f o[4] = {};
  float mrow[8], lrow[8];
#pragma unroll
  for (int r = 0; r < 8; ++r) { mrow[r] = -3.0e38f; lrow[r] = 0.0f; }

  const unsigned char* mp = mask + (size_t)b * LKV;
  _Float16* pb = &plds[wib][0];

  for (int kv0 = 0; kv0 < LKV; kv0 += 32) {
    // ---- scores S = (Q/8) @ K^T, two 16x16 tiles ----
    v8f s[2];
#pragma unroll
    for (int t = 0; t < 2; ++t) {
      const _Float16* krow =
          K + ((size_t)(b * LKV + kv0 + 16 * t + ln)) * CD + h * DK + 16 * grp;
      v8f acc = {};
      acc = wmma16(qa[0], *(const v16h*)(krow), acc);
      acc = wmma16(qa[1], *(const v16h*)(krow + 32), acc);
      const bool pad = mp[kv0 + 16 * t + ln] != 0;  // column = lane%16
#pragma unroll
      for (int r = 0; r < 8; ++r) s[t][r] = pad ? -1.0e9f : acc[r];
    }

    // ---- online softmax (row = r + 8*grp lives across one 16-lane half) ----
#pragma unroll
    for (int r = 0; r < 8; ++r) {
      const float rm   = fmaxf(redmax16(s[0][r]), redmax16(s[1][r]));
      const float mnew = fmaxf(mrow[r], rm);
      const float alpha = __expf(mrow[r] - mnew);
      mrow[r] = mnew;
      const float p0 = __expf(s[0][r] - mnew);
      const float p1 = __expf(s[1][r] - mnew);
      s[0][r] = p0;
      s[1][r] = p1;
      lrow[r] = alpha * lrow[r] + redsum16(p0) + redsum16(p1);
#pragma unroll
      for (int j = 0; j < 4; ++j) o[j][r] *= alpha;
    }

    // ---- P (C/D layout) -> LDS row-major -> A-operand layout ----
#pragma unroll
    for (int t = 0; t < 2; ++t)
#pragma unroll
      for (int r = 0; r < 8; ++r)
        pb[(r + 8 * grp) * 32 + 16 * t + ln] = (_Float16)s[t][r];
    // same-wave DS ops are in-order; compiler inserts s_wait_dscnt for uses
    v16h pa = pack16(*(const v8h*)(pb + ln * 32 + 16 * grp),
                     *(const v8h*)(pb + ln * 32 + 8 + 16 * grp));

    // ---- O += P @ V, B operand contiguous thanks to Vt[b][d][kv] ----
#pragma unroll
    for (int j = 0; j < 4; ++j) {
      const _Float16* vr =
          Vt + ((size_t)(b * CD + h * DK + 16 * j + ln)) * LKV + kv0 + 16 * grp;
      o[j] = wmma16(pa, *(const v16h*)vr, o[j]);
    }
  }

  // ---- normalize and store f16 ----
#pragma unroll
  for (int r = 0; r < 8; ++r) {
    const float inv = 1.0f / lrow[r];
#pragma unroll
    for (int j = 0; j < 4; ++j)
      X[((size_t)(b * LQ + q0 + r + 8 * grp)) * CD + h * DK + 16 * j + ln] =
          (_Float16)(o[j][r] * inv);
  }
}

// ------------------------------ host launch --------------------------------
extern "C" void kernel_launch(void* const* d_in, const int* in_sizes, int n_in,
                              void* d_out, int out_size, void* d_ws,
                              size_t ws_size, hipStream_t stream) {
  (void)in_sizes; (void)n_in; (void)out_size; (void)ws_size;
  const float* query = (const float*)d_in[0];
  const float* key   = (const float*)d_in[1];
  const float* value = (const float*)d_in[2];
  const unsigned char* mask = (const unsigned char*)d_in[3];  // jax bool = u8
  const float* Wq = (const float*)d_in[4];
  const float* bq = (const float*)d_in[5];
  const float* Wk = (const float*)d_in[6];
  const float* bk = (const float*)d_in[7];
  const float* Wv = (const float*)d_in[8];
  const float* bv = (const float*)d_in[9];
  const float* Wo = (const float*)d_in[10];
  const float* bo = (const float*)d_in[11];

  // workspace carve-up (~26 MiB total)
  _Float16* WTq = (_Float16*)d_ws;
  _Float16* WTk = WTq + (size_t)CD * CD;
  _Float16* WTv = WTk + (size_t)CD * CD;
  _Float16* WTo = WTv + (size_t)CD * CD;
  _Float16* Qh  = WTo + (size_t)CD * CD;
  _Float16* Kh  = Qh + (size_t)CB * LQ * CD;
  _Float16* Vt  = Kh + (size_t)CB * LKV * CD;
  _Float16* Xh  = Vt + (size_t)CB * CD * LKV;

  // 1) transpose + convert weights to f16
  wt_kernel<<<(CD * CD) / 256, 256, 0, stream>>>(Wq, WTq);
  wt_kernel<<<(CD * CD) / 256, 256, 0, stream>>>(Wk, WTk);
  wt_kernel<<<(CD * CD) / 256, 256, 0, stream>>>(Wv, WTv);
  wt_kernel<<<(CD * CD) / 256, 256, 0, stream>>>(Wo, WTo);

  // 2) projections: blocks = M/16 (each block = 8 waves, each wave 16x64 tile)
  proj_kernel<0, false><<<(CB * LQ) / 16, 256, 0, stream>>>(query, WTq, bq, Qh,
                                                            CB * LQ);
  proj_kernel<1, false><<<(CB * LKV) / 16, 256, 0, stream>>>(key, WTk, bk, Kh,
                                                             CB * LKV);
  proj_kernel<2, false><<<(CB * LKV) / 16, 256, 0, stream>>>(value, WTv, bv,
                                                             Vt, CB * LKV);

  // 3) attention: B*H*(LQ/16) = 2048 waves -> 256 blocks
  attn_kernel<<<(CB * CH * (LQ / 16)) / 8, 256, 0, stream>>>(Qh, Kh, Vt, mask,
                                                             Xh);

  // 4) output projection, f32 result
  proj_kernel<3, true><<<(CB * LQ) / 16, 256, 0, stream>>>(Xh, WTo, bo, d_out,
                                                           CB * LQ);
}
